// InstanceAwareGeoTransformer_35064113005037
// MI455X (gfx1250) — matrix operand order repaired and checked by reading, not compiled, not run on hardware
//
#include <hip/hip_runtime.h>
#include <hip/hip_fp16.h>
#include <cstdint>

// ---------------------------------------------------------------------------
// InstanceAwareGeoTransformer for MI455X (gfx1250, wave32, WMMA).
// GEMMs on v_wmma_f32_16x16x32_f16. Weight tiles reach LDS two ways:
//  - embed GEMMs: TDM tensor_load_to_lds (one D# per 32-K chunk, TENSORcnt,
//    double-buffered, issued by wave 0 only)
//  - feats GEMM: per-thread global_load_async_to_lds_b128 (ASYNCcnt)
// Geodesic min-plus relaxation LDS-resident (132KB of the 320KB WGP LDS).
// ---------------------------------------------------------------------------

typedef __attribute__((ext_vector_type(16))) _Float16 v16h;
typedef __attribute__((ext_vector_type(8)))  float    v8f;

#define NN          1024
#define MM          1024
#define HH          256
#define CIN         1024
#define KK          64
#define ANGK        3
#define BIN_D_INV   5.0f                  // 1 / 0.2
#define BIN_GEO_INV 10.0f                 // 1 / 0.1
#define BIN_FACT_A  3.8197186342054885f   // 180/(15*pi)
#define GEO_R       0.6f
#define MAXSTEP     32
#define FINF        3.0e38f
#define FINF_THRESH 1.0e30f

// -------------------------- WMMA helpers ----------------------------------

static __device__ __forceinline__ v8f v8f_zero() {
  v8f z;
#pragma unroll
  for (int i = 0; i < 8; ++i) z[i] = 0.0f;
  return z;
}

union Frag16 { v16h v; unsigned int u[8]; };

// A fragment (16x32 f16, ISA 7.12.2): lanes 0-15 row M=lane hold K kb..kb+7
// then kb+16..kb+23; lanes 16-31 row M=lane-16 hold K kb+8.. and kb+24..
static __device__ __forceinline__ v16h load_a_frag(const _Float16* A, int row_base,
                                                   int lda, int kbase) {
  const int lane = threadIdx.x & 31;
  const int row  = row_base + (lane & 15);
  const int kb   = kbase + ((lane < 16) ? 0 : 8);
  const unsigned int* p = (const unsigned int*)(A + row * lda);
  Frag16 f;
#pragma unroll
  for (int i = 0; i < 4; ++i) {
    f.u[i]     = p[(kb >> 1) + i];
    f.u[i + 4] = p[((kb + 16) >> 1) + i];
  }
  return f.v;
}

// B fragment (32x16 f16) from a column-major staged tile Bt[col][32]:
// lanes 0-15 col N=lane hold K 0..7,16..23; lanes 16-31 col N=lane-16 hold K 8..15,24..31.
static __device__ __forceinline__ v16h load_b_frag(const _Float16* Bt, int col_base) {
  const int lane = threadIdx.x & 31;
  const int col  = col_base + (lane & 15);
  const int kb   = (lane < 16) ? 0 : 8;
  const unsigned int* p = (const unsigned int*)(Bt + col * 32 + kb);
  Frag16 f;
#pragma unroll
  for (int i = 0; i < 8; ++i) f.u[i] = p[(i < 4) ? i : i + 4];
  return f.v;
}

static __device__ __forceinline__ v8f wmma16(v16h a, v16h b, v8f c) {
  return __builtin_amdgcn_wmma_f32_16x16x32_f16(false, a, false, b, (short)0, c,
                                                false, false);
}

// -------------------------- async / TDM copy ------------------------------

// Async copy of one 64-byte column slice (4x b128) into LDS; ASYNCcnt-tracked.
static __device__ __forceinline__ void async_copy_col(void* ldsdst, const void* gsrc) {
  unsigned l = (unsigned)(uintptr_t)ldsdst;           // generic->LDS byte addr
  unsigned long long g = (unsigned long long)(uintptr_t)gsrc;
  asm volatile("global_load_async_to_lds_b128 %0, %1, off\n\t"
               "global_load_async_to_lds_b128 %0, %1, off offset:16\n\t"
               "global_load_async_to_lds_b128 %0, %1, off offset:32\n\t"
               "global_load_async_to_lds_b128 %0, %1, off offset:48"
               :: "v"(l), "v"(g) : "memory");
}
static __device__ __forceinline__ void wait_async4() {
  asm volatile("s_wait_asynccnt 4" ::: "memory");
}
static __device__ __forceinline__ void wait_async0() {
  asm volatile("s_wait_asynccnt 0" ::: "memory");
}

// TDM: one Tensor DMA descriptor moving a (tile_dim1=256 rows) x
// (tile_dim0=32 f16) tile with row stride k_elems into contiguous LDS.
// D# per CDNA5 ISA: group0 {count=1, lds_addr,
// global_addr[56:0], type=2}; group1 {data_size=2B, tensor_dim0=k,
// tensor_dim1=256, tile 32x256, dim0_stride=k}. Issued wave-uniform.
static __device__ __forceinline__ void tdm_load_chunk(unsigned lds_addr,
                                                      const void* gsrc,
                                                      unsigned k_elems) {
  unsigned long long ga = (unsigned long long)(uintptr_t)gsrc;
  unsigned g0d0 = 1u;                                        // count=1
  unsigned g0d2 = (unsigned)(ga & 0xFFFFFFFFull);            // addr[31:0]
  unsigned g0d3 = (unsigned)((ga >> 32) & 0x01FFFFFFull) | (2u << 30); // type=2
  unsigned g1d0 = 0x10000u;                                  // data_size=2B
  unsigned g1d1 = (k_elems & 0xFFFFu) << 16;                 // tensor_dim0 lo
  unsigned g1d2 = (k_elems >> 16) | (256u << 16);            // dim0 hi | dim1 lo
  unsigned g1d3 = (32u << 16);                               // tile_dim0=32
  unsigned g1d4 = 256u;                                      // tile_dim1=256
  unsigned g1d5 = k_elems;                                   // dim0_stride lo32
  asm volatile(
      "s_mov_b32 s80, %0\n\t"
      "s_mov_b32 s81, %1\n\t"
      "s_mov_b32 s82, %2\n\t"
      "s_mov_b32 s83, %3\n\t"
      "s_mov_b32 s84, %4\n\t"
      "s_mov_b32 s85, %5\n\t"
      "s_mov_b32 s86, %6\n\t"
      "s_mov_b32 s87, %7\n\t"
      "s_mov_b32 s88, %8\n\t"
      "s_mov_b32 s89, %9\n\t"
      "s_mov_b32 s90, 0\n\t"
      "s_mov_b32 s91, 0\n\t"
      "tensor_load_to_lds s[80:83], s[84:91]"
      :: "s"(g0d0), "s"(lds_addr), "s"(g0d2), "s"(g0d3),
         "s"(g1d0), "s"(g1d1), "s"(g1d2), "s"(g1d3), "s"(g1d4), "s"(g1d5)
      : "memory", "s80", "s81", "s82", "s83", "s84", "s85", "s86", "s87",
        "s88", "s89", "s90", "s91");
}

// One 64xHH @ HHxHH f16 GEMM term accumulated into acc[8]; one wave owns a
// 16-row strip x 128-col half. Wave 0 drives the TDM double-buffer: chunk
// ch+1 DMAs into LDS while chunk ch is multiplied; s_wait_tensorcnt(1)
// releases the older chunk (TDM ops of a wave complete in order).
static __device__ __forceinline__ void gemm_chunks(const _Float16* __restrict__ Alds,
                                                   _Float16* __restrict__ Bt, // [2][HH*32]
                                                   const _Float16* __restrict__ WT,
                                                   v8f acc[8], int strip, int colhalf) {
  const bool wave0 = (threadIdx.x >> 5) == 0;
  if (wave0) tdm_load_chunk((unsigned)(uintptr_t)Bt, WT, HH);
#pragma unroll 1
  for (int ch = 0; ch < 8; ++ch) {
    const _Float16* buf = Bt + (ch & 1) * (HH * 32);
    if (wave0) {
      if (ch < 7) {
        tdm_load_chunk((unsigned)(uintptr_t)(Bt + ((ch + 1) & 1) * (HH * 32)),
                       WT + (ch + 1) * 32, HH);
        __builtin_amdgcn_s_wait_tensorcnt((short)1);
      } else {
        __builtin_amdgcn_s_wait_tensorcnt((short)0);
      }
    }
    __syncthreads();
    v16h a = load_a_frag(Alds, strip * 16, HH, ch * 32);
    const int colbase = colhalf * 128;
    // software-pipelined B fragments: load c+1 while WMMA c executes
    v16h bcur = load_b_frag(buf, colbase);
#pragma unroll
    for (int c = 0; c < 8; ++c) {
      v16h bnxt = (c < 7) ? load_b_frag(buf, colbase + (c + 1) * 16) : bcur;
      acc[c] = wmma16(a, bcur, acc[c]);
      bcur = bnxt;
    }
    __syncthreads();
  }
}

// Build sin_embed rows: A[r][2t]=sin(x_r*div_t), A[r][2t+1]=cos(x_r*div_t).
static __device__ __forceinline__ void stage_embed_A(_Float16* __restrict__ Alds,
                                                     const float* __restrict__ xlds) {
  const int   col    = threadIdx.x;  // 0..255
  const float div    = __expf((float)(col >> 1) * (-9.210340371976184f / 128.0f));
  const bool  is_cos = (col & 1);
#pragma unroll 4
  for (int r = 0; r < 64; ++r) {
    float omg = xlds[r] * div;
    float v   = is_cos ? __cosf(omg) : __sinf(omg);
    Alds[r * HH + col] = (_Float16)v;
  }
}

// -------------------------- small kernels ----------------------------------

// f32 (K x 256) -> f16 transposed (256 x K): WT[col*K + k]
__global__ void cvt_transpose_f16_kernel(const float* __restrict__ src,
                                         _Float16* __restrict__ dst, int K) {
  int t = blockIdx.x * blockDim.x + threadIdx.x;
  if (t < K * HH) {
    int k = t >> 8, c = t & 255;
    dst[c * K + k] = (_Float16)src[t];
  }
}

// One workgroup per query point: brute-force d2 into LDS then 64x min-select.
__global__ void knn_kernel(const float* __restrict__ pts, int n,
                           float* __restrict__ out_dist, int* __restrict__ out_idx) {
  __shared__ __align__(16) float d2s[NN];
  __shared__ float redv[256];
  __shared__ int   redi[256];
  const int q = blockIdx.x, tid = threadIdx.x;
  const float qx = pts[q * 3 + 0], qy = pts[q * 3 + 1], qz = pts[q * 3 + 2];
  for (int j = tid; j < n; j += 256) {
    float dx = pts[j * 3 + 0] - qx;
    float dy = pts[j * 3 + 1] - qy;
    float dz = pts[j * 3 + 2] - qz;
    d2s[j] = dx * dx + dy * dy + dz * dz;
  }
  __syncthreads();
  for (int k = 0; k < KK; ++k) {
    float bv = FINF; int bi = 0x7fffffff;
    for (int j = tid; j < n; j += 256) {
      float v = d2s[j];
      if (v < bv) { bv = v; bi = j; }
    }
    redv[tid] = bv; redi[tid] = bi;
    __syncthreads();
    for (int s = 128; s > 0; s >>= 1) {
      if (tid < s) {
        float ov = redv[tid + s]; int oi = redi[tid + s];
        if (ov < redv[tid] || (ov == redv[tid] && oi < redi[tid])) {
          redv[tid] = ov; redi[tid] = oi;
        }
      }
      __syncthreads();
    }
    if (tid == 0) {
      int wi = redi[0];
      out_dist[q * KK + k] = sqrtf(fmaxf(redv[0], 0.0f));
      out_idx[q * KK + k]  = wi;
      d2s[wi] = FINF;
    }
    __syncthreads();
  }
}

__global__ void refvec_kernel(const float* __restrict__ pts,
                              const int* __restrict__ idx,
                              float* __restrict__ rv, int n) {
  int t = blockIdx.x * blockDim.x + threadIdx.x;
  if (t >= n * ANGK) return;
  int p = t / ANGK, a = t % ANGK;
  int nb = idx[p * KK + 1 + a];
#pragma unroll
  for (int d = 0; d < 3; ++d)
    rv[t * 3 + d] = pts[nb * 3 + d] - pts[p * 3 + d];
}

// Row-independent min-plus relaxation. 16 D-rows per WG live in a double-
// buffered, bank-padded LDS array (132KB -- CDNA5 WGP LDS is 320KB), so the
// per-step idx/w stream (512KB, L2-resident) is amortized over 16 rows.
#define GROWS  16
#define DPITCH 1028   // NN + 4: row padding => D[*][r][ii] spreads across banks
__global__ void geodesic_kernel(const float* __restrict__ dist,
                                const int* __restrict__ idx,
                                float* __restrict__ geo_k_out) {
  __shared__ float D[2][GROWS][DPITCH];
  __shared__ float red[256];
  __shared__ float rmax_s[GROWS];
  const int tid = threadIdx.x;
  const int row0 = blockIdx.x * GROWS;
  for (int e = tid; e < GROWS * NN; e += 256) {
    int r = e >> 10, j = e & (NN - 1);
    D[0][r][j] = (row0 + r == j) ? 0.0f : FINF;
  }
  __syncthreads();
#pragma unroll 1
  for (int step = 0; step < MAXSTEP; ++step) {
    const int cur = step & 1, nxt = cur ^ 1;
#pragma unroll 1
    for (int jj = 0; jj < 4; ++jj) {
      const int j = tid + jj * 256;
      float m[GROWS];
#pragma unroll
      for (int r = 0; r < GROWS; ++r) m[r] = D[cur][r][j];
      for (int k = 0; k < KK; ++k) {
        int   ii = idx[j * KK + k];
        float dv = dist[j * KK + k];
        float wv = (dv <= GEO_R) ? dv : FINF;
#pragma unroll
        for (int r = 0; r < GROWS; ++r)
          m[r] = fminf(m[r], D[cur][r][ii] + wv);
      }
#pragma unroll
      for (int r = 0; r < GROWS; ++r) D[nxt][r][j] = m[r];
    }
    __syncthreads();
  }
  // MAXSTEP is even => final state is in D[0]
#pragma unroll 1
  for (int r = 0; r < GROWS; ++r) {
    float lm = -1.0f;
#pragma unroll
    for (int jj = 0; jj < 4; ++jj) {
      float v = D[0][r][tid + jj * 256];
      if (v < FINF_THRESH) lm = fmaxf(lm, v);
    }
    red[tid] = lm;
    __syncthreads();
    for (int s = 128; s > 0; s >>= 1) {
      if (tid < s) red[tid] = fmaxf(red[tid], red[tid + s]);
      __syncthreads();
    }
    if (tid == 0) rmax_s[r] = red[0];
    __syncthreads();
  }
  for (int e = tid; e < GROWS * KK; e += 256) {
    int r = e / KK, k = e % KK;
    int i = row0 + r;
    int j = idx[i * KK + k];
    float v = D[0][r][j];
    geo_k_out[i * KK + k] = ((v < FINF_THRESH) ? v : rmax_s[r]) * BIN_GEO_INV;
  }
}

// feats (1024x1024) @ W_in + b_in, f16 WMMA; async double-buffered B tiles.
__global__ void feats_gemm_kernel(const float* __restrict__ X,
                                  const _Float16* __restrict__ WT, // [256][1024]
                                  const float* __restrict__ bias,
                                  float* __restrict__ out) {
  __shared__ __align__(16) _Float16 Alds[64 * 32];
  __shared__ __align__(16) _Float16 Bt[2][HH * 32];
  const int tid = threadIdx.x;
  const int wave = tid >> 5, lane = tid & 31;
  const int strip = wave >> 1, colhalf = wave & 1;
  const int row0 = blockIdx.x * 64;
  v8f acc[8];
#pragma unroll
  for (int c = 0; c < 8; ++c) acc[c] = v8f_zero();
  async_copy_col(&Bt[0][tid * 32], WT + tid * CIN);
#pragma unroll 1
  for (int ch = 0; ch < CIN / 32; ++ch) {
    if (ch < CIN / 32 - 1)
      async_copy_col(&Bt[(ch + 1) & 1][tid * 32], WT + tid * CIN + (ch + 1) * 32);
    // stage A chunk f32->f16 (vectorized: 2x float4 per thread)
#pragma unroll
    for (int e = 0; e < 2; ++e) {
      int lin4 = e * 256 + tid;      // 512 float4 per 64x32 chunk
      int r = lin4 >> 3, q = lin4 & 7;
      float4 xv = ((const float4*)(X + (row0 + r) * CIN + ch * 32))[q];
      union { _Float16 h[4]; unsigned long long u; } pk;
      pk.h[0] = (_Float16)xv.x; pk.h[1] = (_Float16)xv.y;
      pk.h[2] = (_Float16)xv.z; pk.h[3] = (_Float16)xv.w;
      *(unsigned long long*)&Alds[r * 32 + q * 4] = pk.u;
    }
    if (ch < CIN / 32 - 1) {
      __builtin_prefetch(X + (row0 + (tid >> 3)) * CIN + (ch + 1) * 32, 0, 1);
      wait_async4();
    } else {
      wait_async0();
    }
    __syncthreads();
    v16h a = load_a_frag(Alds, strip * 16, 32, 0);
    const _Float16* buf = Bt[ch & 1];
    const int colbase = colhalf * 128;
    v16h bcur = load_b_frag(buf, colbase);
#pragma unroll
    for (int c = 0; c < 8; ++c) {
      v16h bnxt = (c < 7) ? load_b_frag(buf, colbase + (c + 1) * 16) : bcur;
      acc[c] = wmma16(a, bcur, acc[c]);
      bcur = bnxt;
    }
    __syncthreads();
  }
  const int rofs = (lane < 16) ? 0 : 8;
  const int cin  = lane & 15;
#pragma unroll
  for (int c = 0; c < 8; ++c) {
    int col = colhalf * 128 + c * 16 + cin;
    float bv = bias[col];
#pragma unroll
    for (int r = 0; r < 8; ++r) {
      int row = row0 + strip * 16 + r + rofs;
      out[row * HH + col] = acc[c][r] + bv;
    }
  }
}

// rge = (sin_embed(dist/BIN_D) @ W_rde + b_rde)
//     + max_a(sin_embed(angle_a) @ W_rae) + b_rae   -- fused, rde never stored.
__global__ void rge_kernel(const float* __restrict__ pts,
                           const float* __restrict__ dist,
                           const int* __restrict__ idx,
                           const float* __restrict__ refvec,
                           const _Float16* __restrict__ WrdeT,
                           const _Float16* __restrict__ WraeT,
                           const float* __restrict__ brde,
                           const float* __restrict__ brae,
                           float* __restrict__ out) {
  __shared__ float xlds[64];
  __shared__ __align__(16) _Float16 Alds[64 * HH];
  __shared__ __align__(16) _Float16 Bt[2][HH * 32];
  const int tid = threadIdx.x;
  const int wave = tid >> 5, lane = tid & 31;
  const int strip = wave >> 1, colhalf = wave & 1;
  const int grow0 = blockIdx.x * 64;

  v8f acc_max[8], t[8];
#pragma unroll 1
  for (int a = 0; a < ANGK; ++a) {
    __syncthreads();
    if (tid < 64) {
      int gr = grow0 + tid;
      int i = gr >> 6, k = gr & 63;
      int nb = idx[i * KK + k];
      float ax = pts[nb * 3 + 0] - pts[i * 3 + 0];
      float ay = pts[nb * 3 + 1] - pts[i * 3 + 1];
      float az = pts[nb * 3 + 2] - pts[i * 3 + 2];
      float rx = refvec[(nb * ANGK + a) * 3 + 0];
      float ry = refvec[(nb * ANGK + a) * 3 + 1];
      float rz = refvec[(nb * ANGK + a) * 3 + 2];
      float cx = ry * az - rz * ay;
      float cy = rz * ax - rx * az;
      float cz = rx * ay - ry * ax;
      float sv = sqrtf(cx * cx + cy * cy + cz * cz);
      float cv = rx * ax + ry * ay + rz * az;
      xlds[tid] = atan2f(sv, cv) * BIN_FACT_A;
    }
    __syncthreads();
    stage_embed_A(Alds, xlds);
    __syncthreads();
#pragma unroll
    for (int c = 0; c < 8; ++c) t[c] = v8f_zero();
    gemm_chunks(Alds, &Bt[0][0], WraeT, t, strip, colhalf);
    if (a == 0) {
#pragma unroll
      for (int c = 0; c < 8; ++c) acc_max[c] = t[c];
    } else {
#pragma unroll
      for (int c = 0; c < 8; ++c)
#pragma unroll
        for (int r = 0; r < 8; ++r)
          acc_max[c][r] = fmaxf(acc_max[c][r], t[c][r]);
    }
  }
  // rde term
  __syncthreads();
  if (tid < 64) xlds[tid] = dist[grow0 + tid] * BIN_D_INV;
  __syncthreads();
  stage_embed_A(Alds, xlds);
  __syncthreads();
#pragma unroll
  for (int c = 0; c < 8; ++c) t[c] = v8f_zero();
  gemm_chunks(Alds, &Bt[0][0], WrdeT, t, strip, colhalf);

  const int rofs = (lane < 16) ? 0 : 8;
  const int cin  = lane & 15;
#pragma unroll
  for (int c = 0; c < 8; ++c) {
    int col = colhalf * 128 + c * 16 + cin;
    float bv = brde[col] + brae[col];
#pragma unroll
    for (int r = 0; r < 8; ++r) {
      int gr = grow0 + strip * 16 + r + rofs;
      out[gr * HH + col] = t[c][r] + acc_max[c][r] + bv;
    }
  }
}

// cross_pe = sin_embed(geo_k) @ W_geo + sin_embed(dist/BIN_D) @ W_rde + biases
// (the two GEMMs share one accumulator).
__global__ void cross_pe_kernel(const float* __restrict__ geok,
                                const float* __restrict__ dist,
                                const _Float16* __restrict__ WgeoT,
                                const _Float16* __restrict__ WrdeT,
                                const float* __restrict__ bgeo,
                                const float* __restrict__ brde,
                                float* __restrict__ out) {
  __shared__ float xlds[64];
  __shared__ __align__(16) _Float16 Alds[64 * HH];
  __shared__ __align__(16) _Float16 Bt[2][HH * 32];
  const int tid = threadIdx.x;
  const int wave = tid >> 5, lane = tid & 31;
  const int strip = wave >> 1, colhalf = wave & 1;
  const int grow0 = blockIdx.x * 64;

  v8f acc[8];
#pragma unroll
  for (int c = 0; c < 8; ++c) acc[c] = v8f_zero();

  if (tid < 64) xlds[tid] = geok[grow0 + tid];
  __syncthreads();
  stage_embed_A(Alds, xlds);
  __syncthreads();
  gemm_chunks(Alds, &Bt[0][0], WgeoT, acc, strip, colhalf);

  __syncthreads();
  if (tid < 64) xlds[tid] = dist[grow0 + tid] * BIN_D_INV;
  __syncthreads();
  stage_embed_A(Alds, xlds);
  __syncthreads();
  gemm_chunks(Alds, &Bt[0][0], WrdeT, acc, strip, colhalf);

  const int rofs = (lane < 16) ? 0 : 8;
  const int cin  = lane & 15;
#pragma unroll
  for (int c = 0; c < 8; ++c) {
    int col = colhalf * 128 + c * 16 + cin;
    float bv = bgeo[col] + brde[col];
#pragma unroll
    for (int r = 0; r < 8; ++r) {
      int gr = grow0 + strip * 16 + r + rofs;
      out[gr * HH + col] = acc[c][r] + bv;
    }
  }
}

__global__ void aux_kernel(const float* __restrict__ rdist, const int* __restrict__ ridx,
                           const float* __restrict__ sdist, const int* __restrict__ sidx,
                           const float* __restrict__ radis_ptr,
                           float* __restrict__ o_ridx, float* __restrict__ o_sidx,
                           float* __restrict__ o_rmsk, float* __restrict__ o_smsk) {
  int t = blockIdx.x * blockDim.x + threadIdx.x;
  if (t >= NN * KK) return;
  float thr = radis_ptr[0] * 0.5f;
  o_ridx[t] = (float)ridx[t];
  o_sidx[t] = (float)sidx[t];
  o_rmsk[t] = (rdist[t] > thr) ? 1.0f : 0.0f;
  o_smsk[t] = (sdist[t] > thr) ? 1.0f : 0.0f;
}

// -------------------------- launcher ---------------------------------------

extern "C" void kernel_launch(void* const* d_in, const int* in_sizes, int n_in,
                              void* d_out, int out_size, void* d_ws, size_t ws_size,
                              hipStream_t stream) {
  (void)in_sizes; (void)n_in; (void)out_size; (void)ws_size;
  const float* ref_pts   = (const float*)d_in[0];
  const float* src_pts   = (const float*)d_in[1];
  const float* ref_feats = (const float*)d_in[2];
  const float* src_feats = (const float*)d_in[3];
  const float* radis     = (const float*)d_in[4];
  const float* W_in      = (const float*)d_in[5];
  const float* b_in      = (const float*)d_in[6];
  const float* W_rde     = (const float*)d_in[7];
  const float* b_rde     = (const float*)d_in[8];
  const float* W_rae     = (const float*)d_in[9];
  const float* b_rae     = (const float*)d_in[10];
  const float* W_geo     = (const float*)d_in[11];
  const float* b_geo     = (const float*)d_in[12];

  char* ws = (char*)d_ws;
  float*    refDist = (float*)   (ws + 0);
  int*      refIdx  = (int*)     (ws + 262144);
  float*    srcDist = (float*)   (ws + 524288);
  int*      srcIdx  = (int*)     (ws + 786432);
  float*    geoK    = (float*)   (ws + 1048576);
  float*    rvRef   = (float*)   (ws + 1310720);
  float*    rvSrc   = (float*)   (ws + 1347584);
  _Float16* WinT    = (_Float16*)(ws + 1384448);   // [256][1024]
  _Float16* WrdeT   = (_Float16*)(ws + 1908736);   // [256][256]
  _Float16* WraeT   = (_Float16*)(ws + 2039808);
  _Float16* WgeoT   = (_Float16*)(ws + 2170880);

  float* out       = (float*)d_out;
  float* o_ref_rge = out + 0;
  float* o_cross   = out + 16777216;
  float* o_src_rge = out + 33554432;
  float* o_ref_f   = out + 50331648;
  float* o_src_f   = out + 50593792;
  float* o_ref_idx = out + 50855936;
  float* o_src_idx = out + 50921472;
  float* o_ref_msk = out + 50987008;
  float* o_src_msk = out + 51052544;

  cvt_transpose_f16_kernel<<<dim3(1024), dim3(256), 0, stream>>>(W_in,  WinT,  CIN);
  cvt_transpose_f16_kernel<<<dim3(256),  dim3(256), 0, stream>>>(W_rde, WrdeT, HH);
  cvt_transpose_f16_kernel<<<dim3(256),  dim3(256), 0, stream>>>(W_rae, WraeT, HH);
  cvt_transpose_f16_kernel<<<dim3(256),  dim3(256), 0, stream>>>(W_geo, WgeoT, HH);

  knn_kernel<<<dim3(NN), dim3(256), 0, stream>>>(ref_pts, NN, refDist, refIdx);
  knn_kernel<<<dim3(MM), dim3(256), 0, stream>>>(src_pts, MM, srcDist, srcIdx);

  refvec_kernel<<<dim3((NN * ANGK + 255) / 256), dim3(256), 0, stream>>>(ref_pts, refIdx, rvRef, NN);
  refvec_kernel<<<dim3((MM * ANGK + 255) / 256), dim3(256), 0, stream>>>(src_pts, srcIdx, rvSrc, MM);

  geodesic_kernel<<<dim3(NN / GROWS), dim3(256), 0, stream>>>(refDist, refIdx, geoK);

  feats_gemm_kernel<<<dim3(NN / 64), dim3(256), 0, stream>>>(ref_feats, WinT, b_in, o_ref_f);
  feats_gemm_kernel<<<dim3(MM / 64), dim3(256), 0, stream>>>(src_feats, WinT, b_in, o_src_f);

  rge_kernel<<<dim3(NN * KK / 64), dim3(256), 0, stream>>>(
      ref_pts, refDist, refIdx, rvRef, WrdeT, WraeT, b_rde, b_rae, o_ref_rge);
  rge_kernel<<<dim3(MM * KK / 64), dim3(256), 0, stream>>>(
      src_pts, srcDist, srcIdx, rvSrc, WrdeT, WraeT, b_rde, b_rae, o_src_rge);

  cross_pe_kernel<<<dim3(NN * KK / 64), dim3(256), 0, stream>>>(
      geoK, refDist, WgeoT, WrdeT, b_geo, b_rde, o_cross);

  aux_kernel<<<dim3(NN * KK / 256), dim3(256), 0, stream>>>(
      refDist, refIdx, srcDist, srcIdx, radis, o_ref_idx, o_src_idx, o_ref_msk, o_src_msk);
}
